// IsingRBM_19112604467252
// MI455X (gfx1250) — matrix operand order repaired and checked by reading, not compiled, not run on hardware
//
#include <hip/hip_runtime.h>
#include <hip/hip_bf16.h>

// CDNA5 / gfx1250: wave32, WMMA 16x16x32 bf16 -> f32 accumulate.
typedef __attribute__((ext_vector_type(16))) __bf16   v16bf;
typedef __attribute__((ext_vector_type(8)))  float    v8f;
typedef __attribute__((ext_vector_type(8)))  unsigned v8u;

#define V_DIM  64
#define H_DIM  128
#define B_DIM  16384
#define K2     4096          // V*V
#define KPAD   4224          // row stride: 4096 (W2) + 64 (W1) + prefetch slack, 132 chunks
#define NCHUNK 130           // 128 order-2 chunks + 2 order-1 chunks

__device__ __forceinline__ unsigned cvt_pk_bf16(float lo, float hi) {
    unsigned d;
    asm("v_cvt_pk_bf16_f32 %0, %1, %2" : "=v"(d) : "v"(lo), "v"(hi));
    return d;
}
__device__ __forceinline__ unsigned pk_mul_bf16(unsigned a, unsigned b) {
    unsigned d;
    asm("v_pk_mul_bf16 %0, %1, %2" : "=v"(d) : "v"(a), "v"(b));
    return d;
}

// ---------- pre-pack: weight_2 (H,V*V) f32 -> bf16 into Wcat[h][0..4095] ----------
__global__ __launch_bounds__(256) void rbm_pack_w2(const float* __restrict__ in,
                                                   __bf16* __restrict__ out) {
    int i = (blockIdx.x * 256 + threadIdx.x) * 8;      // over 128*4096
    int h = i >> 12, k = i & 4095;
    __bf16* o = out + (size_t)h * KPAD + k;
#pragma unroll
    for (int e = 0; e < 8; ++e) o[e] = (__bf16)in[i + e];
}

// ---------- pre-pack: weight_1 (V,H) f32 -> bf16 transposed into Wcat[h][4096..4159] ----------
__global__ __launch_bounds__(256) void rbm_pack_w1(const float* __restrict__ in,
                                                   __bf16* __restrict__ out) {
    int idx = blockIdx.x * 256 + threadIdx.x;          // over 128*64
    int h = idx >> 6, v = idx & 63;
    out[(size_t)h * KPAD + K2 + v] = (__bf16)in[v * H_DIM + h];
}

// ---------- main kernel ----------
// Block = 64 batch rows x 128 hidden. Wave w: mgroup = w>>1 (16 rows),
// colgroup = w&1 (64 cols = 4 accumulators). One A-frag feeds 4 WMMAs.
// Order-1 term folded into the K loop via Wcat + virtual x[:,64] = 2.0.
__global__ __launch_bounds__(256) void rbm_main(const float* __restrict__ x,
                                                const float* __restrict__ hbias,
                                                const __bf16* __restrict__ Wcat,
                                                float* __restrict__ psi) {
    const int tid    = threadIdx.x;
    const int wave   = tid >> 5;
    const int lane   = tid & 31;
    const int half   = lane >> 4;
    const int n      = lane & 15;
    const int mgroup = wave >> 1;        // 0..3
    const int cg     = wave & 1;         // 0..1
    const int row    = mgroup * 16 + n;  // local batch row this lane feeds A from
    const int tile   = blockIdx.x;       // 0..255

    __shared__ float xs[64 * 68];        // padded rows: conflict-free column reads
    __shared__ float prodbuf[8][16];

    // stage x tile (64 x 64 f32) into LDS; entry [r][64] = 2.0 (order-1 scalar trick)
    for (int t = tid; t < 64 * 64; t += 256) {
        int r = t >> 6, c = t & 63;
        xs[r * 68 + c] = x[(size_t)(tile * 64 + r) * V_DIM + c];
    }
    for (int r = tid; r < 64; r += 256) xs[r * 68 + 64] = 2.0f;
    __syncthreads();

    // Pre-pack this lane's j-operands as packed bf16, both chunk parities.
    // A-frag element e -> j = p*32 + e + half*16
    unsigned xjb[2][8];
#pragma unroll
    for (int p = 0; p < 2; ++p)
#pragma unroll
        for (int q = 0; q < 8; ++q) {
            const int e0 = 2 * q, e1 = 2 * q + 1;
            const int j0 = e0 + half * 16;
            const int j1 = e1 + half * 16;
            xjb[p][q] = cvt_pk_bf16(xs[row * 68 + p * 32 + j0],
                                    xs[row * 68 + p * 32 + j1]);
        }

    v8f acc[4];
#pragma unroll
    for (int g = 0; g < 4; ++g) acc[g] = (v8f){0.f,0.f,0.f,0.f,0.f,0.f,0.f,0.f};

    const __bf16* bptr[4];
#pragma unroll
    for (int g = 0; g < 4; ++g)
        bptr[g] = Wcat + (size_t)(cg * 64 + g * 16 + n) * KPAD + half * 16;

    // Ping-pong B buffers in distinct registers: 8 loads always in flight,
    // each buffer waited on a full chunk (4 WMMAs + A build) after issue.
    v16bf bf0[4], bf1[4];
#pragma unroll
    for (int g = 0; g < 4; ++g) bf0[g] = *(const v16bf*)(bptr[g]);
#pragma unroll
    for (int g = 0; g < 4; ++g) bf1[g] = *(const v16bf*)(bptr[g] + 32);

    for (int c = 0; c < NCHUNK; c += 2) {
        // shared broadcast scalar for the parity pair (i = c>>1)
        const float s = 0.5f * xs[row * 68 + (c >> 1)];
        const unsigned sb = cvt_pk_bf16(s, s);

        // ---- chunk c (parity 0) ----
        {
            v8u au;
#pragma unroll
            for (int q = 0; q < 8; ++q) au[q] = pk_mul_bf16(sb, xjb[0][q]);
            const v16bf afrag = __builtin_bit_cast(v16bf, au);
#pragma unroll
            for (int g = 0; g < 4; ++g)
                acc[g] = __builtin_amdgcn_wmma_f32_16x16x32_bf16(
                    false, afrag, false, bf0[g], (short)0, acc[g], false, false);
#pragma unroll
            for (int g = 0; g < 4; ++g)
                bf0[g] = *(const v16bf*)(bptr[g] + (c + 2) * 32);   // <= KPAD-32: in bounds
        }
        // ---- chunk c+1 (parity 1) ----
        {
            v8u au;
#pragma unroll
            for (int q = 0; q < 8; ++q) au[q] = pk_mul_bf16(sb, xjb[1][q]);
            const v16bf afrag = __builtin_bit_cast(v16bf, au);
#pragma unroll
            for (int g = 0; g < 4; ++g)
                acc[g] = __builtin_amdgcn_wmma_f32_16x16x32_bf16(
                    false, afrag, false, bf1[g], (short)0, acc[g], false, false);
#pragma unroll
            for (int g = 0; g < 4; ++g)
                bf1[g] = *(const v16bf*)(bptr[g] + (c + 3) * 32);
        }
    }

    // ---- epilogue: cos(bias + arg), product over this wave's 64 h ----
    float bv[4];
#pragma unroll
    for (int g = 0; g < 4; ++g) bv[g] = hbias[cg * 64 + g * 16 + n];

    float pr[8];
#pragma unroll
    for (int r = 0; r < 8; ++r) {
        float v = __cosf(acc[0][r] + bv[0]);
        v *= __cosf(acc[1][r] + bv[1]);
        v *= __cosf(acc[2][r] + bv[2]);
        v *= __cosf(acc[3][r] + bv[3]);
        // product over the 16 N-lanes of each half (masks < 16 keep halves apart)
        v *= __shfl_xor(v, 1, 32);
        v *= __shfl_xor(v, 2, 32);
        v *= __shfl_xor(v, 4, 32);
        v *= __shfl_xor(v, 8, 32);
        pr[r] = v;
    }

    if (n == 0) {   // lane 0 -> rows 0..7, lane 16 -> rows 8..15 (local to mgroup)
#pragma unroll
        for (int r = 0; r < 8; ++r) prodbuf[wave][half * 8 + r] = pr[r];
    }
    __syncthreads();

    if (tid < 64) { // combine the two column-group waves of each mgroup
        const int mg = tid >> 4, k = tid & 15;
        psi[tile * 64 + tid] = prodbuf[2 * mg][k] * prodbuf[2 * mg + 1][k];
    }
}

extern "C" void kernel_launch(void* const* d_in, const int* in_sizes, int n_in,
                              void* d_out, int out_size, void* d_ws, size_t ws_size,
                              hipStream_t stream) {
    (void)in_sizes; (void)n_in; (void)out_size; (void)ws_size;
    const float* x  = (const float*)d_in[0];
    const float* w1 = (const float*)d_in[1];
    const float* w2 = (const float*)d_in[2];
    const float* hb = (const float*)d_in[3];

    __bf16* WcatB = (__bf16*)d_ws;   // 128 * 4224 * 2 B ~= 1.08 MB

    rbm_pack_w2<<<(H_DIM * K2) / (256 * 8), 256, 0, stream>>>(w2, WcatB);
    rbm_pack_w1<<<(H_DIM * V_DIM) / 256, 256, 0, stream>>>(w1, WcatB);
    rbm_main<<<B_DIM / 64, 256, 0, stream>>>(x, hb, WcatB, (float*)d_out);
}
